// RelationMessagePassing_14096082666191
// MI455X (gfx1250) — matrix operand order repaired and checked by reading, not compiled, not run on hardware
//
#include <hip/hip_runtime.h>
#include <hip/hip_bf16.h>

// Harness contract: only kernel_launch() + __global__ kernels; everything on
// `stream`; scratch only in d_ws; deterministic.

typedef __attribute__((ext_vector_type(16))) __bf16 v16bf;
typedef __attribute__((ext_vector_type(8)))  float  v8f;

union FragBF { unsigned int u[8]; v16bf v; };
union PackBF4 { __bf16 h[4]; unsigned long long ull; };

__device__ inline void atomicMaxFloat(float* addr, float val) {
    // Mixed-sign float max via integer atomics.
    if (val >= 0.0f) atomicMax((int*)addr, __float_as_int(val));
    else             atomicMin((unsigned int*)addr, __float_as_uint(val));
}

// Guaranteed hardware f32 atomic add (no CAS loop), device scope (hits L2).
__device__ inline void atomicAddF32(float* p, float v) {
    asm volatile("global_atomic_add_f32 %0, %1, off scope:SCOPE_DEV"
                 :: "v"(p), "v"(v) : "memory");
}

__global__ void init_ws_kernel(float* exps, float* gmax, int n) {
    int i = blockIdx.x * blockDim.x + threadIdx.x;
    if (i < n) exps[i] = 0.0f;
    if (i == 0) *gmax = -__builtin_inff();
}

// ---------------- per-relation edge-MLP + exp-scatter ----------------
template<int A>
__global__ __launch_bounds__(256) void rel_kernel(
    const float* __restrict__ states,  // [N,64]
    const int*   __restrict__ idx,     // [E*A]
    const float* __restrict__ w1, const float* __restrict__ b1,
    const float* __restrict__ w2, const float* __restrict__ b2,
    float* __restrict__ exps,          // [N,64] accumulates sum(exp(8*out))
    float* __restrict__ gmax,
    int E)                             // number of tuples
{
    constexpr int D  = A * 64;
    constexpr int KT = D / 32;   // K tiles of 32
    constexpr int NT = D / 16;   // N tiles of 16
    constexpr int NW = 8;        // waves per block

    __shared__ __align__(16) __bf16 w1t[D * D];   // transposed: [n][k]
    __shared__ __align__(16) __bf16 w2t[D * D];   // transposed: [n][k]
    __shared__ __align__(16) __bf16 xt[16 * D];   // activations [m][k]
    __shared__ __align__(16) __bf16 ht[16 * D];
    __shared__ int idxt[16 * A];

    const int tid  = threadIdx.x;
    const int wave = tid >> 5;
    const int lane = tid & 31;
    const int lrow = lane & 15;
    const int hi   = lane >> 4;       // lane half

    // Preload weights to LDS, transposed + bf16 (b128 global reads).
    for (int e4 = tid; e4 < D * D / 4; e4 += 256) {
        int e = e4 * 4;
        int k = e / D, n2 = e % D;               // 4 consecutive n2, same k
        float4 a = ((const float4*)w1)[e4];
        w1t[(n2 + 0) * D + k] = (__bf16)a.x;
        w1t[(n2 + 1) * D + k] = (__bf16)a.y;
        w1t[(n2 + 2) * D + k] = (__bf16)a.z;
        w1t[(n2 + 3) * D + k] = (__bf16)a.w;
        float4 b = ((const float4*)w2)[e4];
        w2t[(n2 + 0) * D + k] = (__bf16)b.x;
        w2t[(n2 + 1) * D + k] = (__bf16)b.y;
        w2t[(n2 + 2) * D + k] = (__bf16)b.z;
        w2t[(n2 + 3) * D + k] = (__bf16)b.w;
    }
    __syncthreads();

    const int ntile = (E + 15) >> 4;
    float lmax = -__builtin_inff();

    for (int t = blockIdx.x; t < ntile; t += gridDim.x) {
        const int base = t << 4;

        // Stage index tile (for scatter phase).
        if (tid < 16 * A) {
            int gi = base * A + tid;
            idxt[tid] = (gi < E * A) ? idx[gi] : 0;
        }
        // Gather tuple features -> xt (bf16), 128-bit loads, 64-bit LDS stores.
        for (int v = tid; v < 4 * D; v += 256) {      // 16*D/4 vectors
            int m   = v / (D / 4);
            int col = (v % (D / 4)) * 4;              // never crosses 64-seg
            int tup = base + m;
            float4 f = {0.0f, 0.0f, 0.0f, 0.0f};
            if (tup < E) {
                int node = idx[tup * A + col / 64];
                f = *(const float4*)&states[node * 64 + (col & 63)];
            }
            PackBF4 pk;
            pk.h[0] = (__bf16)f.x; pk.h[1] = (__bf16)f.y;
            pk.h[2] = (__bf16)f.z; pk.h[3] = (__bf16)f.w;
            *(unsigned long long*)&xt[m * D + col] = pk.ull;
        }
        __syncthreads();

        // ---- layer 1: h = relu(x @ w1 + b1) ----
        {
            FragBF af[KT];
            const unsigned* xp = (const unsigned*)&xt[lrow * D];
            #pragma unroll
            for (int kt = 0; kt < KT; ++kt) {
                const unsigned* p = xp + kt * 16 + hi * 4;
                #pragma unroll
                for (int i = 0; i < 4; ++i) { af[kt].u[i] = p[i]; af[kt].u[4 + i] = p[8 + i]; }
            }
            for (int nt = wave; nt < NT; nt += NW) {
                int ng = nt * 16 + lrow;
                float bv = b1[ng];
                v8f c = {bv, bv, bv, bv, bv, bv, bv, bv};
                #pragma unroll
                for (int kt = 0; kt < KT; ++kt) {
                    FragBF bf;
                    const unsigned* bp = (const unsigned*)&w1t[ng * D] + kt * 16 + hi * 4;
                    #pragma unroll
                    for (int i = 0; i < 8; ++i) bf.u[i] = bp[i < 4 ? i : i + 4];
                    c = __builtin_amdgcn_wmma_f32_16x16x32_bf16(
                            false, af[kt].v, false, bf.v, (short)0, c, false, false);
                }
                #pragma unroll
                for (int j = 0; j < 8; ++j) {
                    int m = j + hi * 8;
                    ht[m * D + ng] = (__bf16)fmaxf(c[j], 0.0f);
                }
            }
        }
        __syncthreads();

        // ---- layer 2: out = h @ w2 + b2 ; exp(8*out) scatter ----
        {
            FragBF af[KT];
            const unsigned* hp = (const unsigned*)&ht[lrow * D];
            #pragma unroll
            for (int kt = 0; kt < KT; ++kt) {
                const unsigned* p = hp + kt * 16 + hi * 4;
                #pragma unroll
                for (int i = 0; i < 4; ++i) { af[kt].u[i] = p[i]; af[kt].u[4 + i] = p[8 + i]; }
            }
            for (int nt = wave; nt < NT; nt += NW) {
                int ng = nt * 16 + lrow;
                float bv = b2[ng];
                v8f c = {bv, bv, bv, bv, bv, bv, bv, bv};
                #pragma unroll
                for (int kt = 0; kt < KT; ++kt) {
                    FragBF bf;
                    const unsigned* bp = (const unsigned*)&w2t[ng * D] + kt * 16 + hi * 4;
                    #pragma unroll
                    for (int i = 0; i < 8; ++i) bf.u[i] = bp[i < 4 ? i : i + 4];
                    c = __builtin_amdgcn_wmma_f32_16x16x32_bf16(
                            false, af[kt].v, false, bf.v, (short)0, c, false, false);
                }
                const int part = ng / 64;    // participant slot in tuple
                const int hcol = ng & 63;    // feature
                #pragma unroll
                for (int j = 0; j < 8; ++j) {
                    int m = j + hi * 8;
                    if (base + m < E) {
                        float ov = c[j];
                        lmax = fmaxf(lmax, ov);
                        float ex = __expf(8.0f * ov);
                        int node = idxt[m * A + part];
                        atomicAddF32(&exps[node * 64 + hcol], ex);
                    }
                }
            }
        }
        __syncthreads();
    }

    // wave-level max reduction, one atomic per wave
    #pragma unroll
    for (int off = 16; off > 0; off >>= 1)
        lmax = fmaxf(lmax, __shfl_xor(lmax, off, 32));
    if (lane == 0) atomicMaxFloat(gmax, lmax);
}

// ---------------- final node-update MLP ----------------
__global__ __launch_bounds__(256) void update_kernel(
    const float* __restrict__ states,
    const float* __restrict__ w1, const float* __restrict__ b1,  // [128,128],[128]
    const float* __restrict__ w2, const float* __restrict__ b2,  // [128,64],[64]
    const float* __restrict__ exps, const float* __restrict__ gmaxp,
    float* __restrict__ out, int n)
{
    constexpr int D = 128, DO = 64, KT = D / 32, NW = 8;
    __shared__ __align__(16) __bf16 w1t[D * D];
    __shared__ __align__(16) __bf16 w2t[DO * D];
    __shared__ __align__(16) __bf16 xt[16 * D];
    __shared__ __align__(16) __bf16 ht[16 * D];

    const int tid = threadIdx.x, wave = tid >> 5, lane = tid & 31;
    const int lrow = lane & 15, hi = lane >> 4;

    for (int e4 = tid; e4 < D * D / 4; e4 += 256) {
        int e = e4 * 4, k = e / D, n2 = e % D;
        float4 a = ((const float4*)w1)[e4];
        w1t[(n2 + 0) * D + k] = (__bf16)a.x;
        w1t[(n2 + 1) * D + k] = (__bf16)a.y;
        w1t[(n2 + 2) * D + k] = (__bf16)a.z;
        w1t[(n2 + 3) * D + k] = (__bf16)a.w;
    }
    for (int e4 = tid; e4 < DO * D / 4; e4 += 256) {
        int e = e4 * 4, k = e / DO, n2 = e % DO;
        float4 b = ((const float4*)w2)[e4];
        w2t[(n2 + 0) * D + k] = (__bf16)b.x;
        w2t[(n2 + 1) * D + k] = (__bf16)b.y;
        w2t[(n2 + 2) * D + k] = (__bf16)b.z;
        w2t[(n2 + 3) * D + k] = (__bf16)b.w;
    }
    __syncthreads();

    const float c0 = *gmaxp;
    const float sc = __expf(-8.0f * c0);

    const int ntile = (n + 15) >> 4;
    for (int t = blockIdx.x; t < ntile; t += gridDim.x) {
        const int base = t << 4;
        // build [max_msg | state] tile, 128-bit loads
        for (int v = tid; v < 4 * D; v += 256) {
            int m = v / (D / 4);
            int col = (v % (D / 4)) * 4;
            int node = base + m;
            float4 f = {0.0f, 0.0f, 0.0f, 0.0f};
            if (node < n) {
                if (col < 64) {
                    float4 es = *(const float4*)&exps[node * 64 + col];
                    f.x = __logf(es.x * sc + 1e-16f) * 0.125f + c0;
                    f.y = __logf(es.y * sc + 1e-16f) * 0.125f + c0;
                    f.z = __logf(es.z * sc + 1e-16f) * 0.125f + c0;
                    f.w = __logf(es.w * sc + 1e-16f) * 0.125f + c0;
                } else {
                    f = *(const float4*)&states[node * 64 + (col - 64)];
                }
            }
            PackBF4 pk;
            pk.h[0] = (__bf16)f.x; pk.h[1] = (__bf16)f.y;
            pk.h[2] = (__bf16)f.z; pk.h[3] = (__bf16)f.w;
            *(unsigned long long*)&xt[m * D + col] = pk.ull;
        }
        __syncthreads();

        FragBF af[KT];
        const unsigned* xp = (const unsigned*)&xt[lrow * D];
        #pragma unroll
        for (int kt = 0; kt < KT; ++kt) {
            const unsigned* p = xp + kt * 16 + hi * 4;
            #pragma unroll
            for (int i = 0; i < 4; ++i) { af[kt].u[i] = p[i]; af[kt].u[4 + i] = p[8 + i]; }
        }
        for (int nt = wave; nt < D / 16; nt += NW) {   // 8 N-tiles over 8 waves
            int ng = nt * 16 + lrow;
            float bv = b1[ng];
            v8f c = {bv, bv, bv, bv, bv, bv, bv, bv};
            #pragma unroll
            for (int kt = 0; kt < KT; ++kt) {
                FragBF bf;
                const unsigned* bp = (const unsigned*)&w1t[ng * D] + kt * 16 + hi * 4;
                #pragma unroll
                for (int i = 0; i < 8; ++i) bf.u[i] = bp[i < 4 ? i : i + 4];
                c = __builtin_amdgcn_wmma_f32_16x16x32_bf16(
                        false, af[kt].v, false, bf.v, (short)0, c, false, false);
            }
            #pragma unroll
            for (int j = 0; j < 8; ++j) {
                int m = j + hi * 8;
                ht[m * D + ng] = (__bf16)fmaxf(c[j], 0.0f);
            }
        }
        __syncthreads();

        if (wave < DO / 16) {                          // 4 N-tiles on waves 0-3
            #pragma unroll
            for (int kt = 0; kt < KT; ++kt) {
                const unsigned* p = ((const unsigned*)&ht[lrow * D]) + kt * 16 + hi * 4;
                #pragma unroll
                for (int i = 0; i < 4; ++i) { af[kt].u[i] = p[i]; af[kt].u[4 + i] = p[8 + i]; }
            }
            int ng = wave * 16 + lrow;
            float bv = b2[ng];
            v8f c = {bv, bv, bv, bv, bv, bv, bv, bv};
            #pragma unroll
            for (int kt = 0; kt < KT; ++kt) {
                FragBF bf;
                const unsigned* bp = (const unsigned*)&w2t[ng * D] + kt * 16 + hi * 4;
                #pragma unroll
                for (int i = 0; i < 8; ++i) bf.u[i] = bp[i < 4 ? i : i + 4];
                c = __builtin_amdgcn_wmma_f32_16x16x32_bf16(
                        false, af[kt].v, false, bf.v, (short)0, c, false, false);
            }
            #pragma unroll
            for (int j = 0; j < 8; ++j) {
                int m = j + hi * 8, node = base + m;
                if (node < n) out[node * 64 + ng] = c[j];
            }
        }
        __syncthreads();
    }
}

extern "C" void kernel_launch(void* const* d_in, const int* in_sizes, int n_in,
                              void* d_out, int out_size, void* d_ws, size_t ws_size,
                              hipStream_t stream) {
    const float* states = (const float*)d_in[0];
    const int* idx0 = (const int*)d_in[1];
    const int* idx1 = (const int*)d_in[2];
    const int* idx2 = (const int*)d_in[3];
    const float* r0w1 = (const float*)d_in[4],  *r0b1 = (const float*)d_in[5];
    const float* r0w2 = (const float*)d_in[6],  *r0b2 = (const float*)d_in[7];
    const float* r1w1 = (const float*)d_in[8],  *r1b1 = (const float*)d_in[9];
    const float* r1w2 = (const float*)d_in[10], *r1b2 = (const float*)d_in[11];
    const float* r2w1 = (const float*)d_in[12], *r2b1 = (const float*)d_in[13];
    const float* r2w2 = (const float*)d_in[14], *r2b2 = (const float*)d_in[15];
    const float* uw1  = (const float*)d_in[16], *ub1  = (const float*)d_in[17];
    const float* uw2  = (const float*)d_in[18], *ub2  = (const float*)d_in[19];

    const int n  = in_sizes[0] / 64;
    const int E0 = in_sizes[1] / 1;
    const int E1 = in_sizes[2] / 2;
    const int E2 = in_sizes[3] / 3;

    float* exps = (float*)d_ws;                 // [n*64]
    float* gmax = exps + (size_t)n * 64;        // [1]

    init_ws_kernel<<<dim3((n * 64 + 255) / 256), dim3(256), 0, stream>>>(exps, gmax, n * 64);

    auto grid = [](int tiles) { return dim3(tiles < 2048 ? tiles : 2048); };
    rel_kernel<1><<<grid((E0 + 15) / 16), dim3(256), 0, stream>>>(
        states, idx0, r0w1, r0b1, r0w2, r0b2, exps, gmax, E0);
    rel_kernel<2><<<grid((E1 + 15) / 16), dim3(256), 0, stream>>>(
        states, idx1, r1w1, r1b1, r1w2, r1b2, exps, gmax, E1);
    rel_kernel<3><<<grid((E2 + 15) / 16), dim3(256), 0, stream>>>(
        states, idx2, r2w1, r2b1, r2w2, r2b2, exps, gmax, E2);
    update_kernel<<<grid((n + 15) / 16), dim3(256), 0, stream>>>(
        states, uw1, ub1, uw2, ub2, exps, gmax, (float*)d_out, n);
}
